// TopoGroupDynamicMaskConv2dContextModel_13426067767599
// MI455X (gfx1250) — matrix operand order repaired and checked by reading, not compiled, not run on hardware
//
#include <hip/hip_runtime.h>

// Problem constants (from reference)
#define BATCH 2
#define GRP   2
#define CIN_  192
#define COUT_ 384
#define HH    48
#define WW    48
#define KSZ   5
#define PAD   2
#define LDIM  (HH * WW)            // 2304
#define KDIM  (CIN_ * KSZ * KSZ)   // 4800
#define MDIM  (COUT_ / GRP)        // 192
#define REP   (CIN_ / GRP)         // 96

// Tiling
#define M_TILE 96
#define N_TILE 128
#define C_TILE 64                  // channels per staging round (2 WMMA K-steps)
#define LSTR   72                  // bf16 row stride in LDS (144B pitch)
#define NCHUNK (25 * (CIN_ / C_TILE))   // 75 staging rounds

// Workspace layout (fast path)
#define WT_ELEMS  ((size_t)BATCH * GRP * 25 * MDIM * CIN_)   // 3,686,400 bf16
#define WT_BYTES  (WT_ELEMS * 2)                             // 7,372,800
#define XB_ELEMS  ((size_t)BATCH * CIN_ * LDIM)              // 884,736 bf16
#define XB_BYTES  (XB_ELEMS * 2)

typedef __attribute__((ext_vector_type(16))) __bf16 v16bf;
typedef __attribute__((ext_vector_type(8)))  __bf16 v8bf;
typedef __attribute__((ext_vector_type(8)))  float  v8f;
typedef unsigned __attribute__((ext_vector_type(4))) u32x4;
typedef int      __attribute__((ext_vector_type(4))) i32x4;
typedef int      __attribute__((ext_vector_type(8))) i32x8;

// =====================  Pre-pass kernels (fast path)  =====================

// dkw (f32, [bg][m][c*25+kp]) -> Wt (bf16, [bg][kp][m][c])  : coalesced writes
__global__ __launch_bounds__(256)
void prep_weights(const float* __restrict__ dkw, unsigned short* __restrict__ wt)
{
    const size_t o = (size_t)blockIdx.x * 256 + threadIdx.x;
    if (o >= WT_ELEMS) return;
    const int c  = (int)(o % CIN_);
    const int m  = (int)((o / CIN_) % MDIM);
    const int kp = (int)((o / ((size_t)CIN_ * MDIM)) % 25);
    const int bg = (int)(o / ((size_t)CIN_ * MDIM * 25));
    const float f = dkw[((size_t)bg * MDIM + m) * KDIM + c * 25 + kp];
    union { __bf16 b; unsigned short s; } cv; cv.b = (__bf16)f;
    wt[o] = cv.s;
}

// x (f32) -> bf16, contiguous
__global__ __launch_bounds__(256)
void prep_x(const float* __restrict__ x, unsigned short* __restrict__ xb)
{
    const size_t o = (size_t)blockIdx.x * 256 + threadIdx.x;
    if (o >= XB_ELEMS) return;
    union { __bf16 b; unsigned short s; } cv; cv.b = (__bf16)x[o];
    xb[o] = cv.s;
}

// =====================  Fast main kernel: TDM + WMMA  =====================

__global__ __launch_bounds__(256)
void topo_conv_wmma_tdm(const unsigned short* __restrict__ xbf,
                        const int*   __restrict__ topo,
                        const unsigned short* __restrict__ wt,
                        const float* __restrict__ dkb,
                        float*       __restrict__ out)
{
    __shared__ __align__(16) __bf16 Alds[2][M_TILE * LSTR];   // filled by TDM
    __shared__ __align__(16) __bf16 Blds[2][N_TILE * LSTR];
    __shared__ unsigned MaskLds[N_TILE * 2];

    const int tid  = threadIdx.x;
    const int nblk = blockIdx.x;                 // 0..17
    const int mblk = blockIdx.y;                 // 0..1
    const int bg   = blockIdx.z;                 // 0..3
    const int b = bg >> 1, g = bg & 1;

    const int n0 = nblk * N_TILE;
    const int m0 = mblk * M_TILE;

    // ---- Phase 0: per-pixel topo-causal mask bits ----
    {
        const int p  = tid & (N_TILE - 1);
        const int gi = tid >> 7;
        const int l  = n0 + p;
        const int y  = l / WW, xp = l - y * WW;
        const int tgc = topo[((b * GRP + g) * HH + y) * WW + xp];
        const int* tgi = topo + (size_t)(b * GRP + gi) * HH * WW;
        unsigned bits = 0u;
        #pragma unroll
        for (int kp = 0; kp < 25; ++kp) {
            const int kh = kp / 5, kw = kp - (kp / 5) * 5;
            const int ny = y + kh - PAD, nx = xp + kw - PAD;
            if (ny >= 0 && ny < HH && nx >= 0 && nx < WW) {
                if (tgi[ny * WW + nx] < tgc) bits |= (1u << kp);
            }
        }
        MaskLds[p * 2 + gi] = bits;
    }
    __syncthreads();

    const int lane = tid & 31;
    const int wave = tid >> 5;
    const int wm   = wave & 1;                   // M sub-block (48 rows)
    const int wn   = wave >> 1;                  // N sub-block (32 cols)
    const int lrow = lane & 15;
    const int hi   = lane >> 4;
    const int kbaseA = hi ? 8 : 0;
    const int khalfB = hi ? 16 : 0;

    v8f acc[3][2] = {};

    const unsigned short* xbase = xbf + (size_t)b * CIN_ * LDIM;
    const unsigned long long wtSlab =
        (unsigned long long)(const void*)(wt) + (size_t)bg * 25 * MDIM * CIN_ * 2;

    // B-stage mapping: one pixel, 32 consecutive channels per thread
    const int bp  = tid & (N_TILE - 1);
    const int bc0 = (tid >> 7) * 32;
    const int bl  = n0 + bp;
    const int by  = bl / WW;
    const int bx  = bl - by * WW;
    const unsigned mbits0 = MaskLds[bp * 2 + 0];
    const unsigned mbits1 = MaskLds[bp * 2 + 1];

    // ---- TDM issue: A tile (96 m-rows x 64 channels, bf16) -> LDS buffer ----
    auto tdmA = [&](int chunk, int buf) {
        const int kp = chunk / 3;
        const int c0 = (chunk - kp * 3) * C_TILE;
        // global address of tile start inside the (192c x 192m) tensor for (bg,kp)
        const unsigned long long ga =
            wtSlab + ((size_t)kp * MDIM * CIN_ + (size_t)m0 * CIN_ + c0) * 2;
        // raw LDS byte offset of destination
        typedef __attribute__((address_space(3))) char lds_char;
        const unsigned ldsaddr =
            (unsigned)(unsigned long long)(lds_char*)&Alds[buf][0];
        // D# group 0: count=1, lds_addr, global_addr, type=2
        u32x4 g0;
        g0[0] = 1u;
        g0[1] = ldsaddr;
        g0[2] = (unsigned)ga;
        g0[3] = ((unsigned)(ga >> 32) & 0x01FFFFFFu) | 0x80000000u;
        // D# group 1: data_size=2B, pad 16B every 128B, dims/strides
        i32x8 g1;
        g1[0] = (int)((1u << 16) | (1u << 20) | (4u << 22) | (3u << 25));
        g1[1] = (int)(192u << 16);     // tensor_dim0 = 192 (c)
        g1[2] = (int)(192u << 16);     // tensor_dim1 = 192 (m)
        g1[3] = (int)(64u  << 16);     // tile_dim0 = 64 (c)
        g1[4] = 96;                    // tile_dim1 = 96 (m)
        g1[5] = 192;                   // tensor_dim0_stride = 192
        g1[6] = 0;
        g1[7] = 0;
        i32x4 gz4 = {0, 0, 0, 0};
        i32x8 gz8 = {0, 0, 0, 0, 0, 0, 0, 0};
        __builtin_amdgcn_tensor_load_to_lds(g0, g1, gz4, gz4, gz8, 0);
    };

    // ---- B stage: masked shifted-x (bf16), 64 channels x 128 pixels ----
    auto stageB = [&](int chunk, int buf) {
        const int kp = chunk / 3;
        const int c0 = (chunk - kp * 3) * C_TILE;
        const int kh = kp / 5, kw = kp - (kp / 5) * 5;
        const int ny = by + kh - PAD, nx = bx + kw - PAD;
        const bool inb = ((unsigned)ny < (unsigned)HH) && ((unsigned)nx < (unsigned)WW);
        const int xoff = inb ? (ny * WW + nx) : 0;
        const int cglob = c0 + bc0;
        const unsigned bits = (cglob >= REP) ? mbits1 : mbits0;
        const bool active = inb && ((bits >> kp) & 1u);
        const unsigned keep = active ? 0xFFFFu : 0u;
        const unsigned short* xp = xbase + (size_t)cglob * LDIM + xoff;
        __builtin_prefetch(xp + (size_t)C_TILE * LDIM, 0, 0);   // global_prefetch_b8
        unsigned short* dst = (unsigned short*)&Blds[buf][bp * LSTR + bc0];
        #pragma unroll
        for (int t = 0; t < 32; ++t)
            dst[t] = (unsigned short)(xp[(size_t)t * LDIM] & keep);
    };

    // ---- Compute one round: 2 K-steps x 6 tiles = 12 WMMA per wave ----
    auto compute = [&](int buf) {
        #pragma unroll
        for (int s = 0; s < 2; ++s) {
            v16bf afrag[3], bfrag[2];
            #pragma unroll
            for (int mt = 0; mt < 3; ++mt) {
                const __bf16* ap = &Alds[buf][(wm * 48 + mt * 16 + lrow) * LSTR + s * 32 + kbaseA];
                v8bf lo = *(const v8bf*)(ap);
                v8bf hh = *(const v8bf*)(ap + 16);
                afrag[mt] = __builtin_shufflevector(lo, hh, 0,1,2,3,4,5,6,7,8,9,10,11,12,13,14,15);
            }
            #pragma unroll
            for (int nt = 0; nt < 2; ++nt) {
                const __bf16* bq = &Blds[buf][(wn * 32 + nt * 16 + lrow) * LSTR + s * 32 + kbaseA];
                v8bf lo = *(const v8bf*)(bq);
                v8bf hh = *(const v8bf*)(bq + 16);
                bfrag[nt] = __builtin_shufflevector(lo, hh, 0,1,2,3,4,5,6,7,8,9,10,11,12,13,14,15);
            }
            #pragma unroll
            for (int mt = 0; mt < 3; ++mt)
                #pragma unroll
                for (int nt = 0; nt < 2; ++nt)
                    acc[mt][nt] = __builtin_amdgcn_wmma_f32_16x16x32_bf16(
                        false, afrag[mt], false, bfrag[nt],
                        (short)0, acc[mt][nt], false, false);
        }
    };

    // ---- Pipeline: TDM(A) + VGPR-gather(B) double-buffered, 1 barrier/round ----
    stageB(0, 0);
    if (wave == 0) { tdmA(0, 0); __builtin_amdgcn_s_wait_tensorcnt(0); }
    __syncthreads();
    for (int ch = 0; ch < NCHUNK - 1; ++ch) {
        const int nb = (ch + 1) & 1;
        stageB(ch + 1, nb);
        if (wave == 0) tdmA(ch + 1, nb);         // DMA overlaps this round's compute
        compute(ch & 1);
        if (wave == 0) __builtin_amdgcn_s_wait_tensorcnt(0);
        __syncthreads();
    }
    compute((NCHUNK - 1) & 1);

    // ---- Epilogue: bias + store ----
    const float* biasb = dkb + (size_t)(b * GRP + g) * MDIM;
    #pragma unroll
    for (int mt = 0; mt < 3; ++mt) {
        #pragma unroll
        for (int r = 0; r < 8; ++r) {
            const int m = m0 + wm * 48 + mt * 16 + r + hi * 8;
            const float bias = biasb[m];
            #pragma unroll
            for (int nt = 0; nt < 2; ++nt) {
                const int l = n0 + wn * 32 + nt * 16 + lrow;
                out[((size_t)(b * COUT_ + g * MDIM + m)) * LDIM + l] = acc[mt][nt][r] + bias;
            }
        }
    }
}

// =====================  Fallback kernel (no workspace): round-3 path  =====================

__global__ __launch_bounds__(256)
void topo_conv_wmma_fallback(const float* __restrict__ xin,
                             const int*   __restrict__ topo,
                             const float* __restrict__ dkw,
                             const float* __restrict__ dkb,
                             float*       __restrict__ out)
{
    __shared__ __align__(16) __bf16 Alds[2][M_TILE * LSTR];
    __shared__ __align__(16) __bf16 Blds[2][N_TILE * LSTR];
    __shared__ unsigned MaskLds[N_TILE * 2];

    const int tid  = threadIdx.x;
    const int nblk = blockIdx.x;
    const int mblk = blockIdx.y;
    const int bg   = blockIdx.z;
    const int b = bg >> 1, g = bg & 1;
    const int n0 = nblk * N_TILE;
    const int m0 = mblk * M_TILE;

    {
        const int p  = tid & (N_TILE - 1);
        const int gi = tid >> 7;
        const int l  = n0 + p;
        const int y  = l / WW, xp = l - y * WW;
        const int tgc = topo[((b * GRP + g) * HH + y) * WW + xp];
        const int* tgi = topo + (size_t)(b * GRP + gi) * HH * WW;
        unsigned bits = 0u;
        #pragma unroll
        for (int kp = 0; kp < 25; ++kp) {
            const int kh = kp / 5, kw = kp - (kp / 5) * 5;
            const int ny = y + kh - PAD, nx = xp + kw - PAD;
            if (ny >= 0 && ny < HH && nx >= 0 && nx < WW)
                if (tgi[ny * WW + nx] < tgc) bits |= (1u << kp);
        }
        MaskLds[p * 2 + gi] = bits;
    }
    __syncthreads();

    const int lane = tid & 31;
    const int wave = tid >> 5;
    const int wm   = wave & 1;
    const int wn   = wave >> 1;
    const int lrow = lane & 15;
    const int hi   = lane >> 4;
    const int kbaseA = hi ? 8 : 0;
    const int khalfB = hi ? 16 : 0;

    v8f acc[3][2] = {};
    const float* wbase = dkw + ((size_t)(b * GRP + g) * MDIM + m0) * KDIM;
    const float* xbase = xin + (size_t)b * CIN_ * LDIM;

    const bool aact = tid < 192;
    const int arow  = tid >> 1;
    const int aseg  = (tid & 1) * 32;
    const int bp  = tid & (N_TILE - 1);
    const int bc0 = (tid >> 7) * 32;
    const int bl  = n0 + bp;
    const int by  = bl / WW;
    const int bx  = bl - by * WW;
    const unsigned mbits0 = MaskLds[bp * 2 + 0];
    const unsigned mbits1 = MaskLds[bp * 2 + 1];

    auto stage = [&](int chunk, int buf) {
        const int kp = chunk / 3;
        const int c0 = (chunk - kp * 3) * C_TILE;
        const int kh = kp / 5, kw = kp - (kp / 5) * 5;
        if (aact) {
            const float* src = wbase + (size_t)arow * KDIM + (c0 + aseg) * 25 + kp;
            __bf16* dst = &Alds[buf][arow * LSTR + aseg];
            #pragma unroll
            for (int t = 0; t < 32; ++t) dst[t] = (__bf16)src[t * 25];
        }
        {
            const int ny = by + kh - PAD, nx = bx + kw - PAD;
            const bool inb = ((unsigned)ny < (unsigned)HH) && ((unsigned)nx < (unsigned)WW);
            const int xoff = inb ? (ny * WW + nx) : 0;
            const int cglob = c0 + bc0;
            const unsigned bits = (cglob >= REP) ? mbits1 : mbits0;
            const bool active = inb && ((bits >> kp) & 1u);
            const float scale = active ? 1.f : 0.f;
            const float* xp = xbase + (size_t)cglob * LDIM + xoff;
            __bf16* dst = &Blds[buf][bp * LSTR + bc0];
            #pragma unroll
            for (int t = 0; t < 32; ++t)
                dst[t] = (__bf16)(xp[(size_t)t * LDIM] * scale);
        }
    };

    auto compute = [&](int buf) {
        #pragma unroll
        for (int s = 0; s < 2; ++s) {
            v16bf afrag[3], bfrag[2];
            #pragma unroll
            for (int mt = 0; mt < 3; ++mt) {
                const __bf16* ap = &Alds[buf][(wm * 48 + mt * 16 + lrow) * LSTR + s * 32 + kbaseA];
                v8bf lo = *(const v8bf*)(ap);
                v8bf hh = *(const v8bf*)(ap + 16);
                afrag[mt] = __builtin_shufflevector(lo, hh, 0,1,2,3,4,5,6,7,8,9,10,11,12,13,14,15);
            }
            #pragma unroll
            for (int nt = 0; nt < 2; ++nt) {
                const __bf16* bq = &Blds[buf][(wn * 32 + nt * 16 + lrow) * LSTR + s * 32 + kbaseA];
                v8bf lo = *(const v8bf*)(bq);
                v8bf hh = *(const v8bf*)(bq + 16);
                bfrag[nt] = __builtin_shufflevector(lo, hh, 0,1,2,3,4,5,6,7,8,9,10,11,12,13,14,15);
            }
            #pragma unroll
            for (int mt = 0; mt < 3; ++mt)
                #pragma unroll
                for (int nt = 0; nt < 2; ++nt)
                    acc[mt][nt] = __builtin_amdgcn_wmma_f32_16x16x32_bf16(
                        false, afrag[mt], false, bfrag[nt],
                        (short)0, acc[mt][nt], false, false);
        }
    };

    stage(0, 0);
    __syncthreads();
    for (int ch = 0; ch < NCHUNK - 1; ++ch) {
        stage(ch + 1, (ch + 1) & 1);
        compute(ch & 1);
        __syncthreads();
    }
    compute((NCHUNK - 1) & 1);

    const float* biasb = dkb + (size_t)(b * GRP + g) * MDIM;
    #pragma unroll
    for (int mt = 0; mt < 3; ++mt) {
        #pragma unroll
        for (int r = 0; r < 8; ++r) {
            const int m = m0 + wm * 48 + mt * 16 + r + hi * 8;
            const float bias = biasb[m];
            #pragma unroll
            for (int nt = 0; nt < 2; ++nt) {
                const int l = n0 + wn * 32 + nt * 16 + lrow;
                out[((size_t)(b * COUT_ + g * MDIM + m)) * LDIM + l] = acc[mt][nt][r] + bias;
            }
        }
    }
}

// =====================  Host launcher  =====================

extern "C" void kernel_launch(void* const* d_in, const int* in_sizes, int n_in,
                              void* d_out, int out_size, void* d_ws, size_t ws_size,
                              hipStream_t stream) {
    (void)in_sizes; (void)n_in; (void)out_size;
    const float* x    = (const float*)d_in[0];
    const int*   topo = (const int*)d_in[1];
    const float* dkw  = (const float*)d_in[2];
    const float* dkb  = (const float*)d_in[3];
    float* out = (float*)d_out;

    dim3 grid(LDIM / N_TILE, MDIM / M_TILE, BATCH * GRP);   // (18, 2, 4)
    dim3 block(256);

    if (d_ws && ws_size >= WT_BYTES + XB_BYTES) {
        unsigned short* wt = (unsigned short*)d_ws;
        unsigned short* xb = (unsigned short*)((char*)d_ws + WT_BYTES);
        prep_weights<<<dim3((unsigned)((WT_ELEMS + 255) / 256)), block, 0, stream>>>(dkw, wt);
        prep_x<<<dim3((unsigned)((XB_ELEMS + 255) / 256)), block, 0, stream>>>(x, xb);
        topo_conv_wmma_tdm<<<grid, block, 0, stream>>>(xb, topo, wt, dkb, out);
    } else {
        topo_conv_wmma_fallback<<<grid, block, 0, stream>>>(x, topo, dkw, dkb, out);
    }
}
